// Attention_23373212025198
// MI455X (gfx1250) — matrix operand order repaired and checked by reading, not compiled, not run on hardware
//
#include <hip/hip_runtime.h>

#define B_SZ    4
#define T_SEQ   1024
#define D_MODEL 2048
#define QH_N    32
#define KVH_N   8
#define HD_N    64
#define HD2_N   32
#define KV_D    (KVH_N * HD_N)   // 512

typedef __bf16 bf16_t;
typedef __attribute__((ext_vector_type(16))) __bf16 v16bf;
typedef __attribute__((ext_vector_type(8)))  __bf16 v8bf;
typedef __attribute__((ext_vector_type(4)))  __bf16 v4bf;
typedef __attribute__((ext_vector_type(8)))  float  v8f;
typedef __attribute__((ext_vector_type(8)))  unsigned short u16x8;

static __device__ inline v8f wmma_bf16f32(v16bf a, v16bf b, v8f c) {
  // D = A(16x32 bf16) * B(32x16 bf16) + C(16x16 f32)
  return __builtin_amdgcn_wmma_f32_16x16x32_bf16(false, a, false, b, (short)0, c,
                                                 false, false);
}

// xor-lane exchange via ds_swizzle (single instruction, immediate pattern)
template <int XM>
static __device__ inline float swz_xor(float v) {
#if __has_builtin(__builtin_amdgcn_ds_swizzle)
  return __int_as_float(
      __builtin_amdgcn_ds_swizzle(__float_as_int(v), (XM << 10) | 0x1f));
#else
  return __shfl_xor(v, XM, 32);
#endif
}

// A-fragment (16x32, 16-bit): lane m = lane&15; element i -> k = (lane>>4)*16 + i
static __device__ inline v16bf load_a_frag(const bf16_t* row, int half) {
  return *(const v16bf*)(row + half * 16);
}

// ---------------------------------------------------------------------------
// f32 -> bf16 conversion (vectorized x4)
// ---------------------------------------------------------------------------
__global__ void cvt_f32_bf16(const float* __restrict__ in, bf16_t* __restrict__ out, int n) {
  int i = (blockIdx.x * blockDim.x + threadIdx.x) * 4;
  if (i < n) {
    float4 f = *(const float4*)(in + i);
    v4bf o;
    o[0] = (bf16_t)f.x; o[1] = (bf16_t)f.y; o[2] = (bf16_t)f.z; o[3] = (bf16_t)f.w;
    *(v4bf*)(out + i) = o;
  }
}

// ---------------------------------------------------------------------------
// Tiled bf16 WMMA GEMM: C[M,N] = A[M,K] @ B[K,N]
//   Block tile 128x128, 256 threads (8 waves: 4x2, each 32x64), K-step 32.
//   Compile-time flags: OUT_F32 (f32 vs bf16 store), DO_ROPE (fused RoPE).
// ---------------------------------------------------------------------------
template <int OUT_F32, int DO_ROPE>
__global__ __launch_bounds__(256)
void gemm_bf16_wmma(const bf16_t* __restrict__ A, const bf16_t* __restrict__ Bm,
                    void* __restrict__ Cout, int M, int N, int K,
                    const float* __restrict__ cosp, const float* __restrict__ sinp) {
  __shared__ __align__(64) bf16_t lds_a[128 * 32];   // (m,k) row-major
  __shared__ __align__(64) bf16_t lds_bT[128 * 32];  // (n,k): B transposed

  const int tid  = threadIdx.x;
  const int lane = tid & 31;
  const int half = lane >> 4;
  const int l16  = lane & 15;
  const int wave = tid >> 5;
  const int wm = wave >> 1, wn = wave & 1;     // 4 x 2 wave grid (32 x 64 each)
  const int m0 = blockIdx.y * 128;
  const int n0 = blockIdx.x * 128;

  const v8f zero8 = {0.f, 0.f, 0.f, 0.f, 0.f, 0.f, 0.f, 0.f};
  v8f acc[2][4];
#pragma unroll
  for (int ti = 0; ti < 2; ++ti)
#pragma unroll
    for (int tj = 0; tj < 4; ++tj) acc[ti][tj] = zero8;

  const int arow = tid >> 1, aseg = tid & 1;   // A staging: 128 rows x 2 segs
  const int brp  = tid >> 4, bseg = tid & 15;  // B staging: 16 row-pairs x 16 segs

  for (int k0 = 0; k0 < K; k0 += 32) {
    { // stage A tile: 32B per thread, coalesced
      *(v16bf*)&lds_a[arow * 32 + aseg * 16] =
          *(const v16bf*)&A[(size_t)(m0 + arow) * K + k0 + aseg * 16];
      if (k0 + 32 < K)
        __builtin_prefetch(&A[(size_t)(m0 + arow) * K + k0 + 32 + aseg * 16], 0, 1);
    }
    { // stage B transposed: load a pair of rows, pack 2 bf16 per b32 store
      const u16x8 r0 = *(const u16x8*)&Bm[(size_t)(k0 + 2 * brp) * N + n0 + bseg * 8];
      const u16x8 r1 = *(const u16x8*)&Bm[(size_t)(k0 + 2 * brp + 1) * N + n0 + bseg * 8];
#pragma unroll
      for (int i = 0; i < 8; ++i) {
        unsigned int pk = (unsigned int)r0[i] | ((unsigned int)r1[i] << 16);
        *(unsigned int*)&lds_bT[(bseg * 8 + i) * 32 + 2 * brp] = pk;
      }
      if (k0 + 32 < K)
        __builtin_prefetch(&Bm[(size_t)(k0 + 32 + 2 * brp) * N + n0 + bseg * 8], 0, 1);
    }
    __syncthreads();

    v16bf af[2], bfr[4];
#pragma unroll
    for (int ti = 0; ti < 2; ++ti)
      af[ti] = load_a_frag(&lds_a[(wm * 32 + ti * 16 + l16) * 32], half);
#pragma unroll
    for (int tj = 0; tj < 4; ++tj)  // B-frag: n fixed, k = half*16 + i (contiguous)
      bfr[tj] = *(const v16bf*)&lds_bT[(wn * 64 + tj * 16 + l16) * 32 + half * 16];
#pragma unroll
    for (int ti = 0; ti < 2; ++ti)
#pragma unroll
      for (int tj = 0; tj < 4; ++tj)
        acc[ti][tj] = wmma_bf16f32(af[ti], bfr[tj], acc[ti][tj]);
    __syncthreads();
  }

  // Epilogue: C layout row = r + 8*half, col = lane&15 (per 16x16 tile)
#pragma unroll
  for (int ti = 0; ti < 2; ++ti) {
#pragma unroll
    for (int tj = 0; tj < 4; ++tj) {
#pragma unroll
      for (int r = 0; r < 8; ++r) {
        int gm = m0 + wm * 32 + ti * 16 + r + 8 * half;
        int gn = n0 + wn * 64 + tj * 16 + l16;
        float v = acc[ti][tj][r];
        if (DO_ROPE) {
          float partner = swz_xor<1>(v);          // col gn^1, same row
          int t = gm & (T_SEQ - 1);               // gm = b*T + t
          int h = gn & (HD_N - 1);
          float c = cosp[t * HD2_N + (h >> 1)];
          float s = sinp[t * HD2_N + (h >> 1)];
          v = ((h & 1) == 0) ? (v * c - partner * s) : (partner * s + v * c);
        }
        if (OUT_F32) ((float*)Cout)[(size_t)gm * N + gn] = v;
        else         ((bf16_t*)Cout)[(size_t)gm * N + gn] = (bf16_t)v;
      }
    }
  }
}

// ---------------------------------------------------------------------------
// Flash attention (causal, GQA rep=4): one block = (b, qh, 64 q rows),
// 4 waves x 16 q rows, key chunks of 32, online softmax, all matmuls WMMA.
// ---------------------------------------------------------------------------
__global__ __launch_bounds__(128)
void flash_attn_wmma(const bf16_t* __restrict__ Q, const bf16_t* __restrict__ Kt,
                     const bf16_t* __restrict__ V, bf16_t* __restrict__ O) {
  __shared__ __align__(64) bf16_t lds_k[32 * 64];     // (key, d) row-major
  __shared__ __align__(64) bf16_t lds_vT[64 * 32];    // (d, key): V transposed
  __shared__ __align__(64) bf16_t lds_p[4 * 16 * 32]; // per-wave P tile (m,k)

  const int tid  = threadIdx.x;
  const int lane = tid & 31;
  const int half = lane >> 4;
  const int l16  = lane & 15;
  const int wave = tid >> 5;
  const int q0   = blockIdx.x * 64;
  const int qh   = blockIdx.y;
  const int b    = blockIdx.z;
  const int kvh  = qh >> 2;   // N_REP = 4

  // Q fragments for this wave's 16 rows (held for whole kernel)
  const int qrow = q0 + wave * 16 + l16;   // A-frag: m = lane&15
  const bf16_t* qbase = Q + (size_t)(b * T_SEQ + qrow) * D_MODEL + qh * HD_N;
  v16bf qa0 = load_a_frag(qbase, half);        // d = 0..31
  v16bf qa1 = load_a_frag(qbase + 32, half);   // d = 32..63

  const v8f zero8 = {0.f, 0.f, 0.f, 0.f, 0.f, 0.f, 0.f, 0.f};
  v8f acc[4] = {zero8, zero8, zero8, zero8};   // O: 16 rows x 64 dims
  float mrow[8], lrow[8];
#pragma unroll
  for (int r = 0; r < 8; ++r) { mrow[r] = -1e30f; lrow[r] = 0.f; }

  const int ks  = tid >> 2, kseg = tid & 3;    // K staging
  const int vrp = tid >> 3, vseg = tid & 7;    // V staging: 16 key-pairs x 8 segs

  const int nchunk = (q0 + 64) / 32;
  for (int ch = 0; ch < nchunk; ++ch) {
    const int kc = ch * 32;
    { // stage K chunk (32 x 64), row-major -> QK^T B-frags contiguous
      *(v16bf*)&lds_k[ks * 64 + kseg * 16] =
          *(const v16bf*)&Kt[(size_t)(b * T_SEQ + kc + ks) * KV_D + kvh * HD_N + kseg * 16];
    }
    { // stage V chunk transposed (d, key): pair keys -> packed b32 stores
      const u16x8 r0 =
          *(const u16x8*)&V[(size_t)(b * T_SEQ + kc + 2 * vrp) * KV_D + kvh * HD_N + vseg * 8];
      const u16x8 r1 =
          *(const u16x8*)&V[(size_t)(b * T_SEQ + kc + 2 * vrp + 1) * KV_D + kvh * HD_N + vseg * 8];
#pragma unroll
      for (int i = 0; i < 8; ++i) {
        unsigned int pk = (unsigned int)r0[i] | ((unsigned int)r1[i] << 16);
        *(unsigned int*)&lds_vT[(vseg * 8 + i) * 32 + 2 * vrp] = pk;
      }
    }
    __syncthreads();

    // Scores: S = Q (16x64) @ K^T (64x32) as two 16x16 tiles, K-dim 2x32
    v8f sc[2];
#pragma unroll
    for (int j = 0; j < 2; ++j) {
      v16bf b0 = *(const v16bf*)&lds_k[(j * 16 + l16) * 64 + half * 16];       // d 0..31
      v16bf b1 = *(const v16bf*)&lds_k[(j * 16 + l16) * 64 + 32 + half * 16];  // d 32..63
      v8f c = zero8;
      c = wmma_bf16f32(qa0, b0, c);
      c = wmma_bf16f32(qa1, b1, c);
      sc[j] = c;
    }

    // Online softmax over the 32 keys of this chunk
    float p0f[8], p1f[8], alpha[8];
#pragma unroll
    for (int r = 0; r < 8; ++r) {
      const int qg = q0 + wave * 16 + r + 8 * half;
      float s0 = sc[0][r] * 0.125f;   // 1/sqrt(64)
      float s1 = sc[1][r] * 0.125f;
      if (kc + l16 > qg)      s0 = -1e30f;   // causal mask
      if (kc + 16 + l16 > qg) s1 = -1e30f;
      float mx = fmaxf(s0, s1);
      mx = fmaxf(mx, swz_xor<1>(mx));
      mx = fmaxf(mx, swz_xor<2>(mx));
      mx = fmaxf(mx, swz_xor<4>(mx));
      mx = fmaxf(mx, swz_xor<8>(mx));
      float mnew = fmaxf(mrow[r], mx);
      float p0 = __expf(s0 - mnew);
      float p1 = __expf(s1 - mnew);
      float rs = p0 + p1;
      rs += swz_xor<1>(rs);
      rs += swz_xor<2>(rs);
      rs += swz_xor<4>(rs);
      rs += swz_xor<8>(rs);
      float a = __expf(mrow[r] - mnew);
      alpha[r] = a;
      lrow[r] = lrow[r] * a + rs;
      mrow[r] = mnew;
      p0f[r] = p0; p1f[r] = p1;
    }
#pragma unroll
    for (int dt = 0; dt < 4; ++dt)
#pragma unroll
      for (int r = 0; r < 8; ++r) acc[dt][r] *= alpha[r];

    // C-layout -> A-layout for P via per-wave LDS bounce
    bf16_t* pw = &lds_p[wave * 512];
#pragma unroll
    for (int r = 0; r < 8; ++r) {
      int m = r + 8 * half;
      pw[m * 32 + l16]      = (bf16_t)p0f[r];
      pw[m * 32 + 16 + l16] = (bf16_t)p1f[r];
    }
    __syncthreads();

    v16bf pf = load_a_frag(&pw[l16 * 32], half);
#pragma unroll
    for (int dt = 0; dt < 4; ++dt) {
      v16bf bv = *(const v16bf*)&lds_vT[(dt * 16 + l16) * 32 + half * 16];
      acc[dt] = wmma_bf16f32(pf, bv, acc[dt]);
    }
    __syncthreads();  // protect lds_k/lds_vT before next chunk's staging
  }

  // Normalize and emit bf16 (feeds the final WMMA GEMM with wo)
#pragma unroll
  for (int dt = 0; dt < 4; ++dt)
#pragma unroll
    for (int r = 0; r < 8; ++r) {
      int qg = q0 + wave * 16 + r + 8 * half;
      int d  = dt * 16 + l16;
      float v = acc[dt][r] / lrow[r];
      O[(size_t)(b * T_SEQ + qg) * D_MODEL + qh * HD_N + d] = (bf16_t)v;
    }
}

// ---------------------------------------------------------------------------
extern "C" void kernel_launch(void* const* d_in, const int* in_sizes, int n_in,
                              void* d_out, int out_size, void* d_ws, size_t ws_size,
                              hipStream_t stream) {
  (void)in_sizes; (void)n_in; (void)out_size; (void)ws_size;
  const float* x    = (const float*)d_in[0];
  const float* wq   = (const float*)d_in[1];
  const float* wk   = (const float*)d_in[2];
  const float* wv   = (const float*)d_in[3];
  const float* wo   = (const float*)d_in[4];
  const float* cosp = (const float*)d_in[5];
  const float* sinp = (const float*)d_in[6];
  // d_in[7] = start_pos (0) — causal positions are absolute t.

  const size_t n_x  = (size_t)B_SZ * T_SEQ * D_MODEL;   // 8.39M
  const size_t n_wq = (size_t)D_MODEL * D_MODEL;        // 4.19M
  const size_t n_wk = (size_t)D_MODEL * KV_D;           // 1.05M
  const size_t M    = (size_t)B_SZ * T_SEQ;             // 4096

  bf16_t* p = (bf16_t*)d_ws;
  bf16_t* xb   = p;            p += n_x;
  bf16_t* wqb  = p;            p += n_wq;
  bf16_t* wkb  = p;            p += n_wk;
  bf16_t* wvb  = p;            p += n_wk;
  bf16_t* wob  = p;            p += n_wq;
  bf16_t* Qb   = p;            p += M * D_MODEL;
  bf16_t* Kb   = p;            p += M * KV_D;
  bf16_t* Vb   = p;            p += M * KV_D;
  bf16_t* Attn = p;            p += M * D_MODEL;

  auto cvt = [&](const float* src, bf16_t* dst, size_t n) {
    int blocks = (int)((n / 4 + 255) / 256);
    cvt_f32_bf16<<<blocks, 256, 0, stream>>>(src, dst, (int)n);
  };
  cvt(x,  xb,  n_x);
  cvt(wq, wqb, n_wq);
  cvt(wk, wkb, n_wk);
  cvt(wv, wvb, n_wk);
  cvt(wo, wob, n_wq);

  // Q = rope(x @ wq), K = rope(x @ wk), V = x @ wv   (bf16 outputs)
  gemm_bf16_wmma<0, 1><<<dim3(D_MODEL / 128, M / 128), 256, 0, stream>>>(
      xb, wqb, Qb, (int)M, D_MODEL, D_MODEL, cosp, sinp);
  gemm_bf16_wmma<0, 1><<<dim3(KV_D / 128, M / 128), 256, 0, stream>>>(
      xb, wkb, Kb, (int)M, KV_D, D_MODEL, cosp, sinp);
  gemm_bf16_wmma<0, 0><<<dim3(KV_D / 128, M / 128), 256, 0, stream>>>(
      xb, wvb, Vb, (int)M, KV_D, D_MODEL, cosp, sinp);

  // Fused causal GQA attention
  flash_attn_wmma<<<dim3(T_SEQ / 64, QH_N, B_SZ), 128, 0, stream>>>(Qb, Kb, Vb, Attn);

  // out = Attn @ wo  (f32 output)
  gemm_bf16_wmma<1, 0><<<dim3(D_MODEL / 128, M / 128), 256, 0, stream>>>(
      Attn, wob, (float*)d_out, (int)M, D_MODEL, D_MODEL, cosp, sinp);
}
